// GALE_FA_46359876993418
// MI455X (gfx1250) — matrix operand order repaired and checked by reading, not compiled, not run on hardware
//
#include <hip/hip_runtime.h>
#include <hip/hip_bf16.h>

// ---------------------------------------------------------------------------
// FLARE low-rank attention + cross-attention block for MI455X (gfx1250).
// All matmuls via v_wmma_f32_16x16x32_f16 (fp16 A/B, fp32 accum), wave32.
// GEMM A-tiles staged via GLOBAL_LOAD_ASYNC_TO_LDS_B128 when available.
// ---------------------------------------------------------------------------

typedef _Float16 h16;
typedef __attribute__((ext_vector_type(8)))  _Float16 v8h;
typedef __attribute__((ext_vector_type(16))) _Float16 v16h;
typedef __attribute__((ext_vector_type(8)))  float    v8f;
typedef __attribute__((ext_vector_type(4)))  int      v4i;

#define B_  2
#define N_  16384
#define C_  512
#define H_  8
#define D_  64
#define QG_ 64
#define SC_ 256
#define BN_ (B_ * N_)

#if __has_builtin(__builtin_amdgcn_global_load_async_to_lds_b128)
#define HAS_ASYNC_LDS 1
#endif

// 16-byte global -> LDS copy. Async (ASYNCcnt-tracked) when the toolchain has
// the gfx1250 builtin, otherwise a plain VGPR round-trip.
// Builtin signature (from probe): (v4i AS1* gsrc, v4i AS3* lds, imm off, imm cpol)
__device__ __forceinline__ void copy16_g2l(h16* ldst, const h16* gsrc) {
#ifdef HAS_ASYNC_LDS
  __builtin_amdgcn_global_load_async_to_lds_b128(
      (__attribute__((address_space(1))) v4i*)(h16*)gsrc,
      (__attribute__((address_space(3))) v4i*)ldst, 0, 0);
#else
  *(v8h*)ldst = *(const v8h*)gsrc;
#endif
}

__device__ __forceinline__ void wait_async_lds() {
#ifdef HAS_ASYNC_LDS
#if __has_builtin(__builtin_amdgcn_s_wait_asynccnt)
  __builtin_amdgcn_s_wait_asynccnt(0);
#else
  asm volatile("s_wait_asynccnt 0" ::: "memory");
#endif
#endif
}

// ---------------- WMMA fragment helpers (layouts per CDNA5 ISA 7.12.2) -----

// A fragment: 16x32 fp16, row-major source with leading dim lda (halves).
__device__ __forceinline__ v16h frag_a(const h16* base, int lda) {
  int ln = threadIdx.x & 31;
  const h16* p = base + (ln & 15) * lda + ((ln >> 4) & 1) * 8;
  v8h lo = *(const v8h*)(p);
  v8h hi = *(const v8h*)(p + 16);
  return __builtin_shufflevector(lo, hi, 0,1,2,3,4,5,6,7,8,9,10,11,12,13,14,15);
}

// B fragment: 32x16 fp16 from B^T stored row-major (N x K), ldb in halves.
__device__ __forceinline__ v16h frag_bt(const h16* baseT, int ldb, int koff) {
  int ln = threadIdx.x & 31;
  const h16* p = baseT + (long)(ln & 15) * ldb + koff + ((ln >> 4) & 1) * 8;
  v8h lo = *(const v8h*)(p);
  v8h hi = *(const v8h*)(p + 16);
  return __builtin_shufflevector(lo, hi, 0,1,2,3,4,5,6,7,8,9,10,11,12,13,14,15);
}

__device__ __forceinline__ v8f wmma_f16(v16h a, v16h b, v8f c) {
  return __builtin_amdgcn_wmma_f32_16x16x32_f16(false, a, false, b, (short)0, c,
                                                false, false);
}

// ---------------- Stage 0a: fuse Wp with per-head Wk/Wv/Wcq ----------------
__global__ void k_fuse_w(const float* __restrict__ Wp, const float* __restrict__ bp,
                         const float* __restrict__ Wk, const float* __restrict__ bk,
                         const float* __restrict__ Wv, const float* __restrict__ bv,
                         const float* __restrict__ Wcq, const float* __restrict__ bcq,
                         h16* WfkT, h16* WfvT, h16* WfqT,
                         float* fbk, float* fbv, float* fbq) {
  int idx = blockIdx.x * blockDim.x + threadIdx.x;
  if (idx >= 512 * 512) return;
  int c = idx >> 9;
  int o = idx & 511;
  int h = o >> 6, d = o & 63;
  const float* wpr = Wp + (long)c * 512 + h * 64;
  float sk = 0.f, sv = 0.f, sq = 0.f;
  #pragma unroll 8
  for (int e = 0; e < 64; ++e) {
    float p = wpr[e];
    sk += p * Wk[e * 64 + d];
    sv += p * Wv[e * 64 + d];
    sq += p * Wcq[e * 64 + d];
  }
  WfkT[(long)o * 512 + c] = (h16)sk;
  WfvT[(long)o * 512 + c] = (h16)sv;
  WfqT[(long)o * 512 + c] = (h16)sq;
  if (c == 0) {  // fused biases: bp_h @ W + b
    const float* bpr = bp + h * 64;
    float fk = bk[d], fv = bv[d], fq = bcq[d];
    for (int e = 0; e < 64; ++e) {
      float p = bpr[e];
      fk += p * Wk[e * 64 + d];
      fv += p * Wv[e * 64 + d];
      fq += p * Wcq[e * 64 + d];
    }
    fbk[o] = fk; fbv[o] = fv; fbq[o] = fq;
  }
}

// ---------------- Stage 0b: small casts / transposes -----------------------
__global__ void k_prep(const float* __restrict__ Wck, const float* __restrict__ Wcv,
                       const float* __restrict__ Wo, const float* __restrict__ qg,
                       const float* __restrict__ ctx,
                       h16* WckT, h16* WcvT, h16* WoT, h16* G, h16* ctxh) {
  int idx = blockIdx.x * blockDim.x + threadIdx.x;
  if (idx < 512 * 512) {
    int i = idx >> 9, o = idx & 511;
    WoT[(long)o * 512 + i] = (h16)Wo[idx];     // Wo is (512 in, 512 out)
    ctxh[idx] = (h16)ctx[idx];                 // B*H*SC*CD == 512*512 exactly
  }
  if (idx < 64 * 64) {
    int e = idx >> 6, d = idx & 63;
    WckT[d * 64 + e] = (h16)Wck[idx];          // Wck (64 in, 64 out) -> B^T
    WcvT[d * 64 + e] = (h16)Wcv[idx];
  }
  if (idx < H_ * QG_ * D_) G[idx] = (h16)qg[idx];  // (1,H,64,64) row-major
}

// elementwise fp32 -> fp16 cast (for x)
__global__ void k_cast(const float* __restrict__ src, h16* __restrict__ dst,
                       long n) {
  long i = (long)blockIdx.x * blockDim.x + threadIdx.x;
  if (i < n) dst[i] = (h16)src[i];
}

// ---------------- Generic tiled WMMA GEMM ----------------------------------
// C(M x Ncols) = A(M x Ktot, fp16) @ B (fp16, given as B^T row-major) + bias.
// Block: 256 rows x 64 cols, 8 waves; each wave computes a 32x64 strip
// (2 A-frags share every B-frag -> 8 WMMAs per 32-deep K-chunk).
// mode 1: fp16 head-split row-major  out[((r/RB)*Hp + c>>6)][r%RB][c&63]
// mode 2: fp16 head-split transposed out[((r/RB)*Hp + c>>6)][c&63][r%RB]
// mode 3: fp32 plain row-major       out[r*Ncols + c]
__global__ void k_gemm(const h16* __restrict__ A, int ldA, int M, int Ktot,
                       int Ncols, const h16* __restrict__ Bt,
                       const float* __restrict__ bias, int mode, int RB, int Hp,
                       void* outp) {
  __shared__ __align__(32) h16 As[256 * 32];
  int wave = threadIdx.x >> 5;
  int ln = threadIdx.x & 31;
  int row0 = blockIdx.x * 256;
  int col0 = blockIdx.y * 64;
  v8f acc[2][4] = {};
  for (int kk = 0; kk < Ktot; kk += 32) {
    __syncthreads();
    {   // stage A[row0..+256) x [kk..+32) -> LDS via async b128 copies
      int t = threadIdx.x;
      #pragma unroll
      for (int q = 0; q < 4; ++q) {
        int c16 = q * 256 + t;             // 16-byte chunk id, 1024 total
        int r = c16 >> 2, off = (c16 & 3) * 8;
        copy16_g2l(As + r * 32 + off, A + (long)(row0 + r) * ldA + kk + off);
      }
    }
    if (kk + 32 < Ktot)  // prefetch next B panel slice (global_prefetch_b8)
      __builtin_prefetch(Bt + (long)(col0 + (threadIdx.x & 63)) * Ktot + kk + 32, 0, 1);
    wait_async_lds();
    __syncthreads();
    v16h a0 = frag_a(As + (wave * 32) * 32, 32);
    v16h a1 = frag_a(As + (wave * 32 + 16) * 32, 32);
    #pragma unroll
    for (int t = 0; t < 4; ++t) {
      v16h bf = frag_bt(Bt + (long)(col0 + 16 * t) * Ktot, Ktot, kk);
      acc[0][t] = wmma_f16(a0, bf, acc[0][t]);
      acc[1][t] = wmma_f16(a1, bf, acc[1][t]);
    }
  }
  // epilogue (C/D layout: lane holds col = ln&15, rows r + 8*(ln>>4))
  int hiRow = ((ln >> 4) & 1) * 8;
  int colBase = ln & 15;
  #pragma unroll
  for (int s = 0; s < 2; ++s) {
    #pragma unroll
    for (int t = 0; t < 4; ++t) {
      int c = col0 + 16 * t + colBase;
      float bvv = bias[c];
      #pragma unroll
      for (int r = 0; r < 8; ++r) {
        int rr = row0 + wave * 32 + s * 16 + hiRow + r;
        float val = acc[s][t][r] + bvv;
        if (mode == 3) {
          ((float*)outp)[(long)rr * Ncols + c] = val;
        } else {
          long blk = (long)(rr / RB) * Hp + (c >> 6);
          long idx = (mode == 1)
              ? blk * (long)RB * 64 + (long)(rr % RB) * 64 + (c & 63)
              : blk * (long)RB * 64 + (long)(c & 63) * RB + (rr % RB);
          ((h16*)outp)[idx] = (h16)val;
        }
      }
    }
  }
}

// ---------------- Flash attention (online softmax, scale = 1.0) ------------
// Per wave: one 16-query tile, streams keys in chunks of 32.
// Q: fp16 row-major (rows x 64). K: fp16 row-major (nk x 64) == B^T of scores.
// VT: fp16 transposed values (64 x nk) per (b,h) == B^T of P@V.
// mode 0: store out^T fp16 -> zT[(bh*64 + d)*nq + qrow]
// mode 1: mixed[(b*N + q)*512 + h*64 + d]  = w * out        (w = sigmoid(smix))
// mode 2: mixed[...]                      += (1 - w) * out
__global__ void k_flash(const h16* __restrict__ Q, long qStride, int qPerHead,
                        const h16* __restrict__ Km, long kStride, int kPerHead,
                        const h16* __restrict__ VT,
                        int nq, int nk, int mode,
                        h16* zT, h16* mixed, const float* smix) {
  __shared__ __align__(32) h16 Ps[8][16 * 32];
  int wave = threadIdx.x >> 5;
  int ln = threadIdx.x & 31;
  int bh = blockIdx.x;
  int b = bh >> 3, h = bh & 7;
  int qt = blockIdx.y * 8 + wave;
  if (qt * 16 >= nq) return;               // wave-uniform: EXEC stays all-ones
  const h16* qb = Q + (qPerHead ? (long)h : (long)bh) * qStride + (long)qt * 16 * 64;
  const h16* kb = Km + (kPerHead ? (long)h : (long)bh) * kStride;
  const h16* vtb = VT + (long)bh * 64 * nk;
  h16* ps = Ps[wave];

  v16h qa0 = frag_a(qb, 64);
  v16h qa1 = frag_a(qb + 32, 64);
  v8f acc[4] = {};
  float m[8], l[8];
  #pragma unroll
  for (int r = 0; r < 8; ++r) { m[r] = -1e30f; l[r] = 0.f; }

  for (int kc = 0; kc < nk; kc += 32) {
    // scores S = Q(16x64) . K_chunk^T  -> two 16x16 tiles
    v8f s0 = {}, s1 = {};
    s0 = wmma_f16(qa0, frag_bt(kb + (long)kc * 64, 64, 0), s0);
    s0 = wmma_f16(qa1, frag_bt(kb + (long)kc * 64, 64, 32), s0);
    s1 = wmma_f16(qa0, frag_bt(kb + (long)(kc + 16) * 64, 64, 0), s1);
    s1 = wmma_f16(qa1, frag_bt(kb + (long)(kc + 16) * 64, 64, 32), s1);

    // online softmax: row stats via xor-shuffles over the 16-lane half
    #pragma unroll
    for (int r = 0; r < 8; ++r) {
      float cv = fmaxf(s0[r], s1[r]);
      cv = fmaxf(cv, __shfl_xor(cv, 1, 32));
      cv = fmaxf(cv, __shfl_xor(cv, 2, 32));
      cv = fmaxf(cv, __shfl_xor(cv, 4, 32));
      cv = fmaxf(cv, __shfl_xor(cv, 8, 32));
      float mn = fmaxf(m[r], cv);
      float asc = __expf(m[r] - mn);
      m[r] = mn;
      s0[r] = __expf(s0[r] - mn);
      s1[r] = __expf(s1[r] - mn);
      float rs = s0[r] + s1[r];
      rs += __shfl_xor(rs, 1, 32);
      rs += __shfl_xor(rs, 2, 32);
      rs += __shfl_xor(rs, 4, 32);
      rs += __shfl_xor(rs, 8, 32);
      l[r] = l[r] * asc + rs;
      #pragma unroll
      for (int t = 0; t < 4; ++t) acc[t][r] *= asc;
    }

    // P (C-layout) -> wave-private LDS row-major 16x32 -> A-layout fragment
    {
      int hi = ((ln >> 4) & 1) * 8;
      int cB = ln & 15;
      #pragma unroll
      for (int r = 0; r < 8; ++r) {
        ps[(hi + r) * 32 + cB]      = (h16)s0[r];
        ps[(hi + r) * 32 + 16 + cB] = (h16)s1[r];
      }
    }
    v16h pa = frag_a(ps, 32);
    #pragma unroll
    for (int t = 0; t < 4; ++t) {
      v16h bvf = frag_bt(vtb + (long)(16 * t) * nk, nk, kc);
      acc[t] = wmma_f16(pa, bvf, acc[t]);
    }
  }

  float w = 0.f;
  if (mode) { float sm = *smix; w = 1.f / (1.f + __expf(-sm)); }
  int hi = ((ln >> 4) & 1) * 8;
  int cB = ln & 15;
  #pragma unroll
  for (int r = 0; r < 8; ++r) {
    float inv = 1.f / l[r];
    int qrow = qt * 16 + hi + r;
    #pragma unroll
    for (int t = 0; t < 4; ++t) {
      int d = 16 * t + cB;
      float o = acc[t][r] * inv;
      if (mode == 0) {
        zT[((long)bh * 64 + d) * nq + qrow] = (h16)o;
      } else {
        long idx = ((long)b * N_ + qrow) * (long)C_ + h * 64 + d;
        if (mode == 1) mixed[idx] = (h16)(w * o);
        else           mixed[idx] = (h16)((float)mixed[idx] + (1.f - w) * o);
      }
    }
  }
}

// ---------------------------------------------------------------------------
extern "C" void kernel_launch(void* const* d_in, const int* in_sizes, int n_in,
                              void* d_out, int out_size, void* d_ws, size_t ws_size,
                              hipStream_t stream) {
  const float* x       = (const float*)d_in[0];
  const float* context = (const float*)d_in[1];
  const float* qg      = (const float*)d_in[2];
  const float* Wp  = (const float*)d_in[3];
  const float* bp  = (const float*)d_in[4];
  const float* Wk  = (const float*)d_in[5];
  const float* bk  = (const float*)d_in[6];
  const float* Wv  = (const float*)d_in[7];
  const float* bv  = (const float*)d_in[8];
  const float* Wcq = (const float*)d_in[9];
  const float* bcq = (const float*)d_in[10];
  const float* Wck = (const float*)d_in[11];
  const float* bck = (const float*)d_in[12];
  const float* Wcv = (const float*)d_in[13];
  const float* bcv = (const float*)d_in[14];
  const float* smix = (const float*)d_in[15];
  const float* Wo  = (const float*)d_in[16];
  const float* bo  = (const float*)d_in[17];
  float* out = (float*)d_out;

  // workspace carve-up (256B aligned slabs)
  size_t off = 0;
  char* ws = (char*)d_ws;
  auto alloc = [&](size_t bytes) -> void* {
    void* p = ws + off;
    off += (bytes + 255) & ~(size_t)255;
    return p;
  };
  const size_t BHND = (size_t)B_ * H_ * N_ * D_;          // 16M
  h16*  xh    = (h16*)alloc((size_t)BN_ * C_ * 2);        // x cast to fp16
  h16*  kf    = (h16*)alloc(BHND * 2);                    // k   (b,h,n,d)
  h16*  qf    = (h16*)alloc(BHND * 2);                    // q   (b,h,n,d)
  h16*  vT    = (h16*)alloc(BHND * 2);                    // v^T (b,h,d,n)
  h16*  mixed = (h16*)alloc((size_t)BN_ * C_ * 2);        // gated attn mix
  h16*  ckf   = (h16*)alloc((size_t)B_ * H_ * SC_ * D_ * 2);
  h16*  cvT   = (h16*)alloc((size_t)B_ * H_ * D_ * SC_ * 2);
  h16*  ctxh  = (h16*)alloc((size_t)B_ * H_ * SC_ * D_ * 2);
  h16*  zT    = (h16*)alloc((size_t)B_ * H_ * D_ * QG_ * 2);
  h16*  G     = (h16*)alloc((size_t)H_ * QG_ * D_ * 2);
  h16*  WfkT  = (h16*)alloc(512 * 512 * 2);
  h16*  WfvT  = (h16*)alloc(512 * 512 * 2);
  h16*  WfqT  = (h16*)alloc(512 * 512 * 2);
  h16*  WoT   = (h16*)alloc(512 * 512 * 2);
  h16*  WckT  = (h16*)alloc(64 * 64 * 2);
  h16*  WcvT  = (h16*)alloc(64 * 64 * 2);
  float* fbk  = (float*)alloc(512 * 4);
  float* fbv  = (float*)alloc(512 * 4);
  float* fbq  = (float*)alloc(512 * 4);

  // 0) weight fusion + casts
  k_fuse_w<<<dim3((512 * 512) / 256), dim3(256), 0, stream>>>(
      Wp, bp, Wk, bk, Wv, bv, Wcq, bcq, WfkT, WfvT, WfqT, fbk, fbv, fbq);
  k_prep<<<dim3((512 * 512) / 256), dim3(256), 0, stream>>>(
      Wck, Wcv, Wo, qg, context, WckT, WcvT, WoT, G, ctxh);
  k_cast<<<dim3((BN_ * (long)C_) / 256), dim3(256), 0, stream>>>(
      x, xh, (long)BN_ * C_);

  // 1) fused K/V/Q projections: (BN x 512) @ (512 x 512)
  dim3 gBig(BN_ / 256, 512 / 64);
  k_gemm<<<gBig, dim3(256), 0, stream>>>(xh, 512, BN_, 512, 512, WfkT, fbk,
                                         1, N_, H_, (void*)kf);
  k_gemm<<<gBig, dim3(256), 0, stream>>>(xh, 512, BN_, 512, 512, WfvT, fbv,
                                         2, N_, H_, (void*)vT);
  k_gemm<<<gBig, dim3(256), 0, stream>>>(xh, 512, BN_, 512, 512, WfqT, fbq,
                                         1, N_, H_, (void*)qf);

  // 2) context projections: (B*H*SC x 64) @ (64 x 64)
  dim3 gCtx((B_ * H_ * SC_) / 256, 1);
  k_gemm<<<gCtx, dim3(256), 0, stream>>>(ctxh, 64, B_ * H_ * SC_, 64, 64,
                                         WckT, bck, 1, SC_, 1, (void*)ckf);
  k_gemm<<<gCtx, dim3(256), 0, stream>>>(ctxh, 64, B_ * H_ * SC_, 64, 64,
                                         WcvT, bcv, 2, SC_, 1, (void*)cvT);

  // 3) z = attn(G, k, v): 64 global queries vs N keys   -> zT (b,h,d,qg)
  k_flash<<<dim3(B_ * H_, 1), dim3(256), 0, stream>>>(
      G, (long)QG_ * D_, 1, kf, (long)N_ * D_, 0, vT,
      QG_, N_, 0, zT, nullptr, nullptr);

  // 4) self_attn = attn(k, G, z): N queries vs 64 keys -> mixed = w*self
  k_flash<<<dim3(B_ * H_, N_ / 128), dim3(256), 0, stream>>>(
      kf, (long)N_ * D_, 0, G, (long)QG_ * D_, 1, zT,
      N_, QG_, 1, nullptr, mixed, smix);

  // 5) cross_attn = attn(q, ck, cv): N queries vs 256 keys -> mixed += (1-w)*cross
  k_flash<<<dim3(B_ * H_, N_ / 128), dim3(256), 0, stream>>>(
      qf, (long)N_ * D_, 0, ckf, (long)SC_ * D_, 0, cvT,
      N_, SC_, 2, nullptr, mixed, smix);

  // 6) out = mixed @ Wo + bo  (fp32 out)
  k_gemm<<<gBig, dim3(256), 0, stream>>>(mixed, 512, BN_, 512, 512, WoT, bo,
                                         3, N_, H_, (void*)out);
}
